// M2MHungarianMatcher_77713138253901
// MI455X (gfx1250) — matrix operand order repaired and checked by reading, not compiled, not run on hardware
//
#include <hip/hip_runtime.h>
#include <cstdint>

// Problem constants (from reference setup_inputs)
#define BB 2
#define QQ 300
#define TT 300
#define HH 256
#define WW 256
#define PP 12544     // = 392 * 32 = 49 * 256
#define QP 304       // Q padded to multiple of 16
#define TP 304       // T padded to multiple of 16
#define NBLK 49      // PP / 256  (per-row partial-sum blocks)
#define NSPLIT 4     // split-K factor for the WMMA GEMM
#define KCHUNK (PP / NSPLIT)   // 3136 = 98 * 32

typedef __attribute__((ext_vector_type(16))) _Float16 v16h;
typedef __attribute__((ext_vector_type(8)))  _Float16 v8h;
typedef __attribute__((ext_vector_type(8)))  float    v8f;

// ---------------------------------------------------------------------------
// Bilinear point sampling (matches reference point_sample semantics)
// ---------------------------------------------------------------------------
__device__ __forceinline__ float mask_gather(const float* __restrict__ img, int yi, int xi) {
    bool valid = (xi >= 0) & (xi < WW) & (yi >= 0) & (yi < HH);
    int xc = xi < 0 ? 0 : (xi > WW - 1 ? WW - 1 : xi);
    int yc = yi < 0 ? 0 : (yi > HH - 1 ? HH - 1 : yi);
    float v = img[yc * WW + xc];
    return valid ? v : 0.0f;
}

__device__ __forceinline__ float bilinear(const float* __restrict__ img, float px, float py) {
    float x = px * (float)WW - 0.5f;
    float y = py * (float)HH - 0.5f;
    float x0 = floorf(x), y0 = floorf(y);
    float wx = x - x0, wy = y - y0;
    int xi = (int)x0, yi = (int)y0;
    float v00 = mask_gather(img, yi,     xi);
    float v01 = mask_gather(img, yi,     xi + 1);
    float v10 = mask_gather(img, yi + 1, xi);
    float v11 = mask_gather(img, yi + 1, xi + 1);
    return v00 * (1.f - wy) * (1.f - wx) + v01 * (1.f - wy) * wx
         + v10 * wy * (1.f - wx)         + v11 * wy * wx;
}

// ---------------------------------------------------------------------------
// Kernel 1: sample pred masks -> om (f16), sigmoid(om) (f16),
//           per-block partial sums of softplus(om) and sigmoid(om)
//           (deterministic two-stage reduction; no float atomics).
// grid = (NBLK, QP, BB), block = 256
// ---------------------------------------------------------------------------
__global__ void sample_pred_kernel(const float* __restrict__ pred_masks,
                                   const float* __restrict__ pts,
                                   _Float16* __restrict__ om16,
                                   _Float16* __restrict__ s16,
                                   float* __restrict__ pneg,   // [BB*QP][NBLK]
                                   float* __restrict__ ps) {   // [BB*QP][NBLK]
    __shared__ float sdN[256];
    __shared__ float sdS[256];
    int p = blockIdx.x * 256 + threadIdx.x;
    int q = blockIdx.y;
    int b = blockIdx.z;
    float om = 0.f, neg = 0.f, s = 0.f;
    if (q < QQ) {
        float px = pts[((size_t)b * PP + p) * 2 + 0];
        float py = pts[((size_t)b * PP + p) * 2 + 1];
        const float* img = pred_masks + ((size_t)b * QQ + q) * (size_t)(HH * WW);
        om  = bilinear(img, px, py);
        neg = fmaxf(om, 0.f) + log1pf(expf(-fabsf(om)));   // softplus(om), stable
        s   = 1.f / (1.f + expf(-om));                     // sigmoid(om)
    }
    size_t idx = ((size_t)b * QP + q) * (size_t)PP + p;
    om16[idx] = (_Float16)om;
    s16[idx]  = (_Float16)s;
    sdN[threadIdx.x] = neg;
    sdS[threadIdx.x] = s;
    __syncthreads();
    for (int st = 128; st > 0; st >>= 1) {
        if (threadIdx.x < st) {
            sdN[threadIdx.x] += sdN[threadIdx.x + st];
            sdS[threadIdx.x] += sdS[threadIdx.x + st];
        }
        __syncthreads();
    }
    if (threadIdx.x == 0) {
        size_t row = (size_t)b * QP + q;
        pneg[row * NBLK + blockIdx.x] = sdN[0];
        ps[row * NBLK + blockIdx.x]   = sdS[0];
    }
}

// ---------------------------------------------------------------------------
// Kernel 2: sample tgt masks -> tm (f16), per-block partial column sums.
// grid = (NBLK, TP, BB), block = 256
// ---------------------------------------------------------------------------
__global__ void sample_tgt_kernel(const float* __restrict__ tgt_masks,
                                  const float* __restrict__ pts,
                                  _Float16* __restrict__ tm16,
                                  float* __restrict__ ptm) {  // [BB*TP][NBLK]
    __shared__ float sdT[256];
    int p = blockIdx.x * 256 + threadIdx.x;
    int t = blockIdx.y;
    int b = blockIdx.z;
    float tm = 0.f;
    if (t < TT) {
        float px = pts[((size_t)b * PP + p) * 2 + 0];
        float py = pts[((size_t)b * PP + p) * 2 + 1];
        const float* img = tgt_masks + ((size_t)b * TT + t) * (size_t)(HH * WW);
        tm = bilinear(img, px, py);
    }
    size_t idx = ((size_t)b * TP + t) * (size_t)PP + p;
    tm16[idx] = (_Float16)tm;
    sdT[threadIdx.x] = tm;
    __syncthreads();
    for (int st = 128; st > 0; st >>= 1) {
        if (threadIdx.x < st) sdT[threadIdx.x] += sdT[threadIdx.x + st];
        __syncthreads();
    }
    if (threadIdx.x == 0) ptm[((size_t)b * TP + t) * NBLK + blockIdx.x] = sdT[0];
}

// ---------------------------------------------------------------------------
// Deterministic second-stage reduction: [rows][NBLK] -> [rows]
// grid = rows, block = 64
// ---------------------------------------------------------------------------
__global__ void row_reduce_kernel(const float* __restrict__ partial,
                                  float* __restrict__ outsum) {
    __shared__ float sd[64];
    int row = blockIdx.x;
    int i = threadIdx.x;
    sd[i] = (i < NBLK) ? partial[(size_t)row * NBLK + i] : 0.f;
    __syncthreads();
    for (int st = 32; st > 0; st >>= 1) {
        if (i < st) sd[i] += sd[i + st];
        __syncthreads();
    }
    if (i == 0) outsum[row] = sd[0];
}

// ---------------------------------------------------------------------------
// Kernel 3: split-K WMMA GEMM, one wave per (16x16 tile, K-split).
//   accOm = om @ tm^T ; accS = sigmoid(om) @ tm^T   (share the B fragment)
// Fragment assembly follows CDNA5 ISA 16-bit A(16x32)/B(32x16) VGPR layouts:
//   A, lanes 0-15:  halves 0-7 = K 0..7,  halves 8-15 = K 16..23 (M = lane)
//   A, lanes 16-31: halves 0-7 = K 8..15, halves 8-15 = K 24..31 (M = lane-16)
//   B, lanes 0-15:  halves 0-7 = K 0..7,  halves 8-15 = K 16..23 (N = lane)
//   B, lanes 16-31: halves 0-7 = K 8..15, halves 8-15 = K 24..31 (N = lane-16)
//   C/D f32: VGPR r -> M = r + 8*(lane>=16), N = lane&15
// grid = (QP/16, TP/16, BB*NSPLIT), block = 32 (one wave, EXEC all ones)
// 19*19*8 = 2888 waves ≈ 3 waves per SIMD32 -> hides L2 latency.
// ---------------------------------------------------------------------------
__global__ void cost_gemm_kernel(const _Float16* __restrict__ om16,
                                 const _Float16* __restrict__ s16,
                                 const _Float16* __restrict__ tm16,
                                 float* __restrict__ partials_om, // [NSPLIT*BB][QP][TP]
                                 float* __restrict__ partials_s) {
    const int lane  = threadIdx.x;          // 0..31
    const int hi    = lane >> 4;            // lane group
    const int n     = lane & 15;
    const int qt    = blockIdx.x;
    const int tt    = blockIdx.y;
    const int b     = blockIdx.z / NSPLIT;
    const int split = blockIdx.z % NSPLIT;
    const int k0    = split * KCHUNK;

    const _Float16* Aom = om16 + (size_t)b * QP * PP
                        + (size_t)(qt * 16 + n) * PP + (hi ? 8 : 0) + k0;
    const _Float16* As  = s16  + (size_t)b * QP * PP
                        + (size_t)(qt * 16 + n) * PP + (hi ? 8 : 0) + k0;
    const _Float16* Bp  = tm16 + (size_t)b * TP * PP
                        + (size_t)(tt * 16 + n) * PP + (hi ? 8 : 0) + k0;

    v8f accOm = {};
    v8f accS  = {};

    for (int k = 0; k < KCHUNK; k += 32) {
        v8h a0 = *(const v8h*)(Aom + k);
        v8h a1 = *(const v8h*)(Aom + k + 16);
        v8h c0 = *(const v8h*)(As  + k);
        v8h c1 = *(const v8h*)(As  + k + 16);
        v8h b0 = *(const v8h*)(Bp  + k);
        v8h b1 = *(const v8h*)(Bp  + k + 16);
        v16h afrag = __builtin_shufflevector(a0, a1, 0,1,2,3,4,5,6,7,8,9,10,11,12,13,14,15);
        v16h sfrag = __builtin_shufflevector(c0, c1, 0,1,2,3,4,5,6,7,8,9,10,11,12,13,14,15);
        v16h bfrag = __builtin_shufflevector(b0, b1, 0,1,2,3,4,5,6,7,8,9,10,11,12,13,14,15);
        // (neg_a, A, neg_b, B, c_mod, C, reuse_a, reuse_b)
        accOm = __builtin_amdgcn_wmma_f32_16x16x32_f16(false, afrag, false, bfrag,
                                                       (short)0, accOm, false, false);
        accS  = __builtin_amdgcn_wmma_f32_16x16x32_f16(false, sfrag, false, bfrag,
                                                       (short)0, accS,  false, false);
    }

    // Store the 16x16 f32 partial tile (C/D layout -> row-major scratch).
    size_t planebase = (size_t)(split * BB + b) * QP * TP;
    for (int r = 0; r < 8; ++r) {
        size_t o = planebase + (size_t)(qt * 16 + r + hi * 8) * TP + (tt * 16 + n);
        partials_om[o] = accOm[r];
        partials_s[o]  = accS[r];
    }
}

// ---------------------------------------------------------------------------
// Kernel 4: epilogue — ordered split-K reduction + mask/dice/bbox/giou costs.
// grid = ceil(BB*QQ*TT / 256), block = 256
// ---------------------------------------------------------------------------
__global__ void epilogue_kernel(const float* __restrict__ partials_om,
                                const float* __restrict__ partials_s,
                                const float* __restrict__ sum_neg,  // [BB*QP]
                                const float* __restrict__ sum_s,    // [BB*QP]
                                const float* __restrict__ sum_tm,   // [BB*TP]
                                const float* __restrict__ pred_boxes,
                                const float* __restrict__ tgt_boxes,
                                float* __restrict__ out) {
    int idx = blockIdx.x * 256 + threadIdx.x;
    if (idx >= BB * QQ * TT) return;
    int t = idx % TT;
    int q = (idx / TT) % QQ;
    int b = idx / (QQ * TT);

    float omv = 0.f, sv = 0.f;
    for (int sp = 0; sp < NSPLIT; ++sp) {           // fixed order: deterministic
        size_t o = (size_t)(sp * BB + b) * QP * TP + (size_t)q * TP + t;
        omv += partials_om[o];
        sv  += partials_s[o];
    }

    // cost_mask = (rowsum(softplus(om)) - om@tm) / P   (algebraic identity:
    //   pos@tm + neg@(1-tm) = rowsum(neg) - om@tm, since softplus(-x)-softplus(x)=-x)
    float cm = (sum_neg[b * QP + q] - omv) * (1.0f / (float)PP);
    // cost_dice = 1 - (2*s@tm + 1) / (rowsum(s) + colsum(tm) + 1)
    float cd = 1.f - (2.f * sv + 1.f) / (sum_s[b * QP + q] + sum_tm[b * TP + t] + 1.f);

    const float* pb = pred_boxes + ((size_t)b * QQ + q) * 4;
    const float* tb = tgt_boxes  + ((size_t)b * TT + t) * 4;
    float pcx = pb[0], pcy = pb[1], pw = pb[2], ph = pb[3];
    float tcx = tb[0], tcy = tb[1], tw = tb[2], th = tb[3];
    float cb = fabsf(pcx - tcx) + fabsf(pcy - tcy) + fabsf(pw - tw) + fabsf(ph - th);

    float px0 = pcx - 0.5f * pw, py0 = pcy - 0.5f * ph;
    float px1 = pcx + 0.5f * pw, py1 = pcy + 0.5f * ph;
    float tx0 = tcx - 0.5f * tw, ty0 = tcy - 0.5f * th;
    float tx1 = tcx + 0.5f * tw, ty1 = tcy + 0.5f * th;
    float areaA = (px1 - px0) * (py1 - py0);
    float areaB = (tx1 - tx0) * (ty1 - ty0);
    float iw = fmaxf(fminf(px1, tx1) - fmaxf(px0, tx0), 0.f);
    float ih = fmaxf(fminf(py1, ty1) - fmaxf(py0, ty0), 0.f);
    float inter = iw * ih;
    float uni = areaA + areaB - inter;
    float iou = inter / uni;
    float ew = fmaxf(fmaxf(px1, tx1) - fminf(px0, tx0), 0.f);
    float eh = fmaxf(fmaxf(py1, ty1) - fminf(py0, ty0), 0.f);
    float areaE = ew * eh;
    float giou = iou - (areaE - uni) / areaE;

    out[idx] = 5.f * cm + 5.f * cd + 5.f * cb - 2.f * giou;
}

// ---------------------------------------------------------------------------
// Host launcher
// ---------------------------------------------------------------------------
extern "C" void kernel_launch(void* const* d_in, const int* in_sizes, int n_in,
                              void* d_out, int out_size, void* d_ws, size_t ws_size,
                              hipStream_t stream) {
    (void)in_sizes; (void)n_in; (void)out_size; (void)ws_size;

    const float* pred_masks = (const float*)d_in[0];   // (B,Q,H,W)
    const float* tgt_masks  = (const float*)d_in[1];   // (B,T,H,W)
    const float* pred_boxes = (const float*)d_in[2];   // (B,Q,4)
    const float* tgt_boxes  = (const float*)d_in[3];   // (B,T,4)
    const float* pts        = (const float*)d_in[4];   // (B,P,2)
    float* out = (float*)d_out;                        // (B,Q,T)

    // Workspace layout, each region 256B-aligned.
    char* ws = (char*)d_ws;
    size_t off = 0;
    auto alloc = [&](size_t bytes) {
        size_t o = off;
        off = (off + bytes + 255) & ~(size_t)255;
        return o;
    };
    const size_t szA  = (size_t)BB * QP * PP * sizeof(_Float16);   // 15,253,504
    const size_t szPR = (size_t)BB * QP * NBLK * sizeof(float);    // per-row partials
    const size_t szSM = (size_t)BB * QP * sizeof(float);           // reduced sums
    const size_t szPT = (size_t)NSPLIT * BB * QP * TP * sizeof(float); // GEMM partials

    _Float16* om16 = (_Float16*)(ws + alloc(szA));
    _Float16* s16  = (_Float16*)(ws + alloc(szA));
    _Float16* tm16 = (_Float16*)(ws + alloc(szA));
    float* pneg    = (float*)(ws + alloc(szPR));
    float* ps      = (float*)(ws + alloc(szPR));
    float* ptm     = (float*)(ws + alloc(szPR));
    float* sum_neg = (float*)(ws + alloc(szSM));
    float* sum_s   = (float*)(ws + alloc(szSM));
    float* sum_tm  = (float*)(ws + alloc(szSM));
    float* part_om = (float*)(ws + alloc(szPT));
    float* part_s  = (float*)(ws + alloc(szPT));

    // Stage 1: point sampling + per-block partial sums.
    dim3 blkS(256);
    sample_pred_kernel<<<dim3(NBLK, QP, BB), blkS, 0, stream>>>(
        pred_masks, pts, om16, s16, pneg, ps);
    sample_tgt_kernel<<<dim3(NBLK, TP, BB), blkS, 0, stream>>>(
        tgt_masks, pts, tm16, ptm);

    // Stage 2: deterministic second-stage row reductions.
    row_reduce_kernel<<<dim3(BB * QP), dim3(64), 0, stream>>>(pneg, sum_neg);
    row_reduce_kernel<<<dim3(BB * QP), dim3(64), 0, stream>>>(ps,   sum_s);
    row_reduce_kernel<<<dim3(BB * TP), dim3(64), 0, stream>>>(ptm,  sum_tm);

    // Stage 3: split-K WMMA GEMM (2888 waves for latency hiding).
    cost_gemm_kernel<<<dim3(QP / 16, TP / 16, BB * NSPLIT), dim3(32), 0, stream>>>(
        om16, s16, tm16, part_om, part_s);

    // Stage 4: ordered split-K reduction + cost epilogue.
    int nout = BB * QQ * TT;
    epilogue_kernel<<<dim3((nout + 255) / 256), blkS, 0, stream>>>(
        part_om, part_s, sum_neg, sum_s, sum_tm, pred_boxes, tgt_boxes, out);
}
